// GatedGraphConv_51951924412557
// MI455X (gfx1250) — matrix operand order repaired and checked by reading, not compiled, not run on hardware
//
#include <hip/hip_runtime.h>
#include <hip/hip_bf16.h>

// GatedGraphConv for MI455X (gfx1250, wave32).
// B=4, N=512, CIN=FOUT=64.  BN = B*N = 2048 rows.
//
// ws layout (floats):
//   h  [2048*64]   @ 0
//   Uh [2048*64]   @ 131072
//   Vh [2048*64]   @ 262144
//   li [2048]      @ 393216
//   lj [2048]      @ 395264
//   s  [512]       @ 397312      (adj row sums; einsum 'bijf,jk->bif'
//                                 == sum_j nf[b,i,j,f] * rowsum(adj)[j])
//   pk [2048*32 float4] @ 397824 (packed hot-loop operands, 1 MB)
// total ~2.6 MB of d_ws.

#define NNODES 512
#define FD     64
#define BN     2048
#define NEG_SLOPE 0.2f
#define LOG2E  1.4426950408889634f

typedef __attribute__((ext_vector_type(16))) _Float16 v16h;
typedef __attribute__((ext_vector_type(8)))  float    v8f;

// ---- WMMA operand loaders (ISA 7.12.2 layouts, wave32) --------------------
// A 16x32 f16: lanes 0-15 -> row M=lane, elems 0..7 = K k0+0..7, elems 8..15 =
// K k0+16..23; lanes 16-31 -> row M=lane-16, K ranges shifted by +8.
__device__ inline v16h load_A16x32(const float* __restrict__ X, int ld,
                                   int row0, int k0, int lane) {
  const int half = lane >> 4;
  const int m    = lane & 15;
  const float* p = X + (size_t)(row0 + m) * ld + k0 + half * 8;
  v16h a;
#pragma unroll
  for (int e = 0; e < 8; ++e) a[e]     = (_Float16)p[e];
#pragma unroll
  for (int e = 0; e < 8; ++e) a[8 + e] = (_Float16)p[16 + e];
  return a;
}

// B 32x16 f16 where B[k][n] = W[n][k] (W row-major FOUTxCIN -> computes X@W^T):
// lanes 0-15 -> col N=n0+lane, elems 0..7 = K k0+0..7, elems 8..15 = K k0+16..23;
// lanes 16-31 -> K ranges shifted by +8.
__device__ inline v16h load_BT32x16(const float* __restrict__ W, int ld,
                                    int n0, int k0, int lane) {
  const int half = lane >> 4;
  const int n    = n0 + (lane & 15);
  const float* p = W + (size_t)n * ld + k0 + half * 8;
  v16h b;
#pragma unroll
  for (int e = 0; e < 8; ++e) b[e]     = (_Float16)p[e];
#pragma unroll
  for (int e = 0; e < 8; ++e) b[8 + e] = (_Float16)p[16 + e];
  return b;
}

__device__ inline v8f wmma_f32_f16(v16h a, v16h b, v8f c) {
  return __builtin_amdgcn_wmma_f32_16x16x32_f16(false, a, false, b,
                                                (short)0, c, false, false);
}

// C 16x16 f32: elem r -> (M = r + 8*half, N = lane&15) at tile (row0, n0).
__device__ inline void store_C(float* __restrict__ out, int ld, int row0,
                               int n0, int lane, v8f c, float bias) {
  const int half = lane >> 4;
  const int n    = n0 + (lane & 15);
#pragma unroll
  for (int r = 0; r < 8; ++r) {
    const int row = row0 + r + half * 8;
    out[(size_t)row * ld + n] = c[r] + bias;
  }
}

// ---- Kernel 1: h = x @ W^T + Wb  (2048x64 = 128 row-tiles, 1 wave/tile) ----
__global__ __launch_bounds__(128) void k_gemm_h(const float* __restrict__ x,
                                                const float* __restrict__ Ww,
                                                const float* __restrict__ Wb,
                                                float* __restrict__ h) {
  const int lane = threadIdx.x & 31;
  const int wv   = threadIdx.x >> 5;
  const int row0 = (blockIdx.x * 4 + wv) * 16;

  v16h a0 = load_A16x32(x, FD, row0, 0, lane);
  v16h a1 = load_A16x32(x, FD, row0, 32, lane);
#pragma unroll
  for (int n0 = 0; n0 < FD; n0 += 16) {
    v8f c = {};
    c = wmma_f32_f16(a0, load_BT32x16(Ww, FD, n0, 0, lane), c);
    c = wmma_f32_f16(a1, load_BT32x16(Ww, FD, n0, 32, lane), c);
    store_C(h, FD, row0, n0, lane, c, Wb[n0 + (lane & 15)]);
  }
}

// ---- Kernel 2: Uh = h@U^T+Ub, Vh = h@V^T+Vb (reuse A-tile of h) -----------
__global__ __launch_bounds__(128) void k_gemm_uv(const float* __restrict__ h,
                                                 const float* __restrict__ Uw,
                                                 const float* __restrict__ Ub,
                                                 const float* __restrict__ Vw,
                                                 const float* __restrict__ Vb,
                                                 float* __restrict__ Uh,
                                                 float* __restrict__ Vh) {
  const int lane = threadIdx.x & 31;
  const int wv   = threadIdx.x >> 5;
  const int row0 = (blockIdx.x * 4 + wv) * 16;

  v16h a0 = load_A16x32(h, FD, row0, 0, lane);
  v16h a1 = load_A16x32(h, FD, row0, 32, lane);
#pragma unroll
  for (int n0 = 0; n0 < FD; n0 += 16) {
    v8f cu = {}, cv = {};
    cu = wmma_f32_f16(a0, load_BT32x16(Uw, FD, n0, 0, lane), cu);
    cu = wmma_f32_f16(a1, load_BT32x16(Uw, FD, n0, 32, lane), cu);
    cv = wmma_f32_f16(a0, load_BT32x16(Vw, FD, n0, 0, lane), cv);
    cv = wmma_f32_f16(a1, load_BT32x16(Vw, FD, n0, 32, lane), cv);
    store_C(Uh, FD, row0, n0, lane, cu, Ub[n0 + (lane & 15)]);
    store_C(Vh, FD, row0, n0, lane, cv, Vb[n0 + (lane & 15)]);
  }
}

// ---- Kernel 3: li = h.a_i, lj = h.a_j (one wave per row) ------------------
__global__ __launch_bounds__(256) void k_dots(const float* __restrict__ h,
                                              const float* __restrict__ attw,
                                              float* __restrict__ li,
                                              float* __restrict__ lj) {
  const int lane = threadIdx.x & 31;
  const int row  = blockIdx.x * 8 + (threadIdx.x >> 5);
  const float h0 = h[(size_t)row * FD + lane];
  const float h1 = h[(size_t)row * FD + lane + 32];
  float vi = h0 * attw[lane]      + h1 * attw[lane + 32];
  float vj = h0 * attw[64 + lane] + h1 * attw[96 + lane];
#pragma unroll
  for (int m = 16; m > 0; m >>= 1) {
    vi += __shfl_xor(vi, m, 32);
    vj += __shfl_xor(vj, m, 32);
  }
  if (lane == 0) { li[row] = vi; lj[row] = vj; }
}

// ---- Kernel 4: s[j] = sum_k adj[j,k] (one wave per row) -------------------
__global__ __launch_bounds__(256) void k_rowsum(const float* __restrict__ adj,
                                                float* __restrict__ s) {
  const int lane = threadIdx.x & 31;
  const int j    = blockIdx.x * 8 + (threadIdx.x >> 5);
  float acc = 0.f;
#pragma unroll
  for (int t = 0; t < 16; ++t) acc += adj[(size_t)j * NNODES + lane + t * 32];
#pragma unroll
  for (int m = 16; m > 0; m >>= 1) acc += __shfl_xor(acc, m, 32);
  if (lane == 0) s[j] = acc;
}

// ---- Kernel 5: pack hot-loop operands -------------------------------------
// pk[row*32+lane] = { LOG2E*Vh[row,lane], LOG2E*Vh[row,lane+32],
//                     h[row,lane],        h[row,lane+32] }
// One b128 load in the aggregation loop instead of 4 b32 loads; the LOG2E
// pre-scale feeds v_exp_f32 (base-2) with a single subtract per channel.
__global__ __launch_bounds__(256) void k_pack(const float* __restrict__ h,
                                              const float* __restrict__ Vh,
                                              float4* __restrict__ pk) {
  const int idx  = blockIdx.x * 256 + threadIdx.x;  // 0..65535
  const int row  = idx >> 5;
  const int lane = idx & 31;
  const float v0 = Vh[(size_t)row * FD + lane];
  const float v1 = Vh[(size_t)row * FD + lane + 32];
  const float h0 = h[(size_t)row * FD + lane];
  const float h1 = h[(size_t)row * FD + lane + 32];
  pk[idx] = make_float4(LOG2E * v0, LOG2E * v1, h0, h1);
}

// ---- Kernel 6: softmax + gated aggregation + LayerNorm --------------------
// One wave32 per (b,i) row; lane owns channels f=lane and f=lane+32.
__global__ __launch_bounds__(256) void k_aggr(const float4* __restrict__ pk,
                                              const float* __restrict__ Uh,
                                              const float* __restrict__ li,
                                              const float* __restrict__ lj,
                                              const float* __restrict__ s,
                                              const float* __restrict__ attb,
                                              const float* __restrict__ lng,
                                              const float* __restrict__ lnb,
                                              float* __restrict__ out) {
  __shared__ float q[8][NNODES];  // 16 KB: per-wave exp(logit)*s[j]
  const int lane = threadIdx.x & 31;
  const int wv   = threadIdx.x >> 5;
  const int p    = blockIdx.x * 8 + wv;   // 0..2047
  const int b    = p >> 9;
  const int i    = p & 511;

  const float4* __restrict__ pkb = pk + (size_t)b * NNODES * 32;
  const float*  __restrict__ ljb = lj + (size_t)b * NNODES;
  const float   c0 = li[(size_t)b * NNODES + i] + attb[0];

  // Pass 1: leaky-relu logits, running max (16 j's per lane).
  float myl[16];
  float lmax = -3.402823466e38f;
#pragma unroll
  for (int t = 0; t < 16; ++t) {
    const int j = lane + t * 32;
    float l = c0 + ljb[j];
    l = l > 0.f ? l : NEG_SLOPE * l;
    myl[t] = l;
    lmax = fmaxf(lmax, l);
  }
#pragma unroll
  for (int m = 16; m > 0; m >>= 1) lmax = fmaxf(lmax, __shfl_xor(lmax, m, 32));

  // Pass 2: exp, denom; stash q[j] = exp(l-lmax) * s[j] in LDS.
  float sum = 0.f;
#pragma unroll
  for (int t = 0; t < 16; ++t) {
    const int j = lane + t * 32;
    const float e = __builtin_amdgcn_exp2f((myl[t] - lmax) * LOG2E);
    sum += e;
    q[wv][j] = e * s[j];
  }
#pragma unroll
  for (int m = 16; m > 0; m >>= 1) sum += __shfl_xor(sum, m, 32);
  const float inv = __builtin_amdgcn_rcpf(sum);
  __syncthreads();  // cross-lane LDS handoff (uniform across all 8 waves)

  // Pass 3: sum_j q[j] * sigmoid(Uh[i,f]+Vh[j,f]) * h[j,f].
  // sigmoid(u+v) = rcp(1 + exp2(-(LOG2E*u + LOG2E*v))); nu = -LOG2E*u hoisted,
  // pk.x/.y already carry LOG2E*v -> one v_sub + v_exp + v_add + v_rcp each.
  const float nu0 = -LOG2E * Uh[((size_t)b * NNODES + i) * FD + lane];
  const float nu1 = -LOG2E * Uh[((size_t)b * NNODES + i) * FD + lane + 32];
  float a0 = 0.f, a1 = 0.f;
  for (int j = 0; j < NNODES; ++j) {
    const float  qj = q[wv][j];
    const float4 w  = pkb[(size_t)j * 32 + lane];
    const float g0 = __builtin_amdgcn_rcpf(1.f + __builtin_amdgcn_exp2f(nu0 - w.x));
    const float g1 = __builtin_amdgcn_rcpf(1.f + __builtin_amdgcn_exp2f(nu1 - w.y));
    a0 = fmaf(qj * g0, w.z, a0);
    a1 = fmaf(qj * g1, w.w, a1);
  }
  a0 *= inv;
  a1 *= inv;

  // LayerNorm over the 64 channels.
  float sm = a0 + a1, sq = a0 * a0 + a1 * a1;
#pragma unroll
  for (int m = 16; m > 0; m >>= 1) {
    sm += __shfl_xor(sm, m, 32);
    sq += __shfl_xor(sq, m, 32);
  }
  const float mu  = sm * (1.f / 64.f);
  const float var = sq * (1.f / 64.f) - mu * mu;
  const float rs  = __builtin_amdgcn_rsqf(var + 1e-5f);
  const size_t base = ((size_t)b * NNODES + i) * FD;
  out[base + lane]      = (a0 - mu) * rs * lng[lane]      + lnb[lane];
  out[base + lane + 32] = (a1 - mu) * rs * lng[lane + 32] + lnb[lane + 32];
}

extern "C" void kernel_launch(void* const* d_in, const int* in_sizes, int n_in,
                              void* d_out, int out_size, void* d_ws, size_t ws_size,
                              hipStream_t stream) {
  (void)in_sizes; (void)n_in; (void)out_size; (void)ws_size;
  const float* x    = (const float*)d_in[0];
  const float* adj  = (const float*)d_in[1];
  const float* Ww   = (const float*)d_in[2];
  const float* Wb   = (const float*)d_in[3];
  const float* Uw   = (const float*)d_in[4];
  const float* Ub   = (const float*)d_in[5];
  const float* Vw   = (const float*)d_in[6];
  const float* Vb   = (const float*)d_in[7];
  const float* attw = (const float*)d_in[8];
  const float* attb = (const float*)d_in[9];
  const float* lng  = (const float*)d_in[10];
  const float* lnb  = (const float*)d_in[11];
  float* out = (float*)d_out;

  float*  ws = (float*)d_ws;
  float*  h  = ws;
  float*  Uh = ws + 131072;
  float*  Vh = ws + 262144;
  float*  li = ws + 393216;
  float*  lj = ws + 395264;
  float*  s  = ws + 397312;
  float4* pk = (float4*)(ws + 397824);  // 16B-aligned (397824*4 % 16 == 0)

  k_gemm_h <<<dim3(32),  dim3(128), 0, stream>>>(x, Ww, Wb, h);
  k_gemm_uv<<<dim3(32),  dim3(128), 0, stream>>>(h, Uw, Ub, Vw, Vb, Uh, Vh);
  k_dots   <<<dim3(256), dim3(256), 0, stream>>>(h, attw, li, lj);
  k_rowsum <<<dim3(64),  dim3(256), 0, stream>>>(adj, s);
  k_pack   <<<dim3(256), dim3(256), 0, stream>>>(h, Vh, pk);
  k_aggr   <<<dim3(256), dim3(256), 0, stream>>>(pk, Uh, li, lj, s,
                                                 attb, lng, lnb, out);
}
